// GatedFFN_17506286698976
// MI455X (gfx1250) — matrix operand order repaired and checked by reading, not compiled, not run on hardware
//
#include <hip/hip_runtime.h>
#include <hip/hip_bf16.h>

// GatedFFN with hard top-1 tile routing (forward gate == one-hot argmax):
//   phase 1: fp32 gate logits + argmax, one-hot gate output, per-tile row compaction
//   phase 1.5 (if ws allows): convert Wup (full) and Wdn (routed quarter, compacted)
//             to f16 in d_ws -> halves L2 weight bytes, removes cvt from hot loop
//   phase 2: per-tile fused GEMM chain X@Wup_t^T -> relu -> @Wdown_t^T with
//            v_wmma_f32_16x16x32_f16; 64 rows/block, 4-way M register blocking.

typedef _Float16 v16h __attribute__((ext_vector_type(16)));
typedef _Float16 h8   __attribute__((ext_vector_type(8)));
typedef _Float16 h4   __attribute__((ext_vector_type(4)));
typedef float    v8f  __attribute__((ext_vector_type(8)));

#define C_DIM   1024
#define DFF     4096
#define NTILES  4
#define BM      64     // rows per block (4 row-groups of 16)
#define XSTR    1032   // 1024 + 8 f16 pad -> conflict-free A-fragment LDS reads

// LDS carve (bytes)
#define XLDS_BYTES (BM * XSTR * 2)              // 132096
#define HLDS_OFF   XLDS_BYTES
#define OLDS_OFF   0                            // olds aliases dead xlds
#define RIDX_OFF   (2 * XLDS_BYTES)
#define LDS_TOTAL  (2 * XLDS_BYTES + BM * 4)    // 264448

// ws layout (bytes): counts | perm | WupH (f16) | WdnH (f16, compacted)
#define WS_PERM_OFF   256
#define WUP_ELEMS     (DFF * C_DIM)                       // 4M
#define WDN_ELEMS     (NTILES * 256 * C_DIM)              // 1M (routed quarter)

// ---------- helpers -------------------------------------------------------

// A-fragment (16x32 f16, MxK). Lane L (half = L>>4, m = L&15) holds row m with
// 16 contiguous K values: K = kb + half*16 + i.
__device__ __forceinline__ v16h load_a_frag(const _Float16* __restrict__ rowp,
                                            int kb, int half) {
  h8 a0 = *(const h8*)(rowp + kb + half * 16);
  h8 a1 = *(const h8*)(rowp + kb + half * 16 + 8);
  return __builtin_shufflevector(a0, a1, 0, 1, 2, 3, 4, 5, 6, 7,
                                 8, 9, 10, 11, 12, 13, 14, 15);
}

// B-fragment (32x16 f16, KxN), B[k][n] = W[n][k]. Lane L (n = L&15, half = L>>4)
// holds 16 contiguous K values: K = kb + half*16 + i.
__device__ __forceinline__ v16h load_b_frag_f32(const float* __restrict__ p) {
  v16h b;
#pragma unroll
  for (int i = 0; i < 16; i += 4) {
    float4 f = *(const float4*)(p + i);
    b[i + 0] = (_Float16)f.x;
    b[i + 1] = (_Float16)f.y;
    b[i + 2] = (_Float16)f.z;
    b[i + 3] = (_Float16)f.w;
  }
  return b;
}

__device__ __forceinline__ v16h load_b_frag_f16(const _Float16* __restrict__ p) {
  h8 b0 = *(const h8*)(p);
  h8 b1 = *(const h8*)(p + 8);
  return __builtin_shufflevector(b0, b1, 0, 1, 2, 3, 4, 5, 6, 7,
                                 8, 9, 10, 11, 12, 13, 14, 15);
}

#define WMMA_F16(A, B, Cacc) \
  __builtin_amdgcn_wmma_f32_16x16x32_f16(false, (A), false, (B), (short)0, (Cacc), false, false)

// ---------- phase 0: reset routing counters -------------------------------

__global__ void zero_counts_kernel(int* __restrict__ counts) {
  if (threadIdx.x < NTILES) counts[threadIdx.x] = 0;
}

// ---------- phase 1: gate logits, argmax, one-hot, compaction --------------

__global__ void gate_kernel(const float* __restrict__ x,
                            const float* __restrict__ Wg,
                            const float* __restrict__ bg,
                            float* __restrict__ gate_out,
                            int* __restrict__ counts,
                            int* __restrict__ perm, int N) {
  __shared__ float wgl[NTILES * C_DIM];
  int tid = threadIdx.x;
  for (int i = tid; i < NTILES * C_DIM; i += 256) wgl[i] = Wg[i];
  __syncthreads();

  int wave = tid >> 5, lane = tid & 31;
  int n = blockIdx.x * 8 + wave;   // one wave32 per row
  if (n >= N) return;

  const float* xr = x + (size_t)n * C_DIM;
  float a0 = 0.f, a1 = 0.f, a2 = 0.f, a3 = 0.f;
  for (int k = lane; k < C_DIM; k += 32) {
    float xv = xr[k];
    a0 += xv * wgl[k];
    a1 += xv * wgl[C_DIM + k];
    a2 += xv * wgl[2 * C_DIM + k];
    a3 += xv * wgl[3 * C_DIM + k];
  }
#pragma unroll
  for (int off = 16; off > 0; off >>= 1) {
    a0 += __shfl_xor(a0, off, 32);
    a1 += __shfl_xor(a1, off, 32);
    a2 += __shfl_xor(a2, off, 32);
    a3 += __shfl_xor(a3, off, 32);
  }
  if (lane == 0) {
    float l0 = a0 + bg[0], l1 = a1 + bg[1], l2 = a2 + bg[2], l3 = a3 + bg[3];
    int bt = 0;
    float bv = l0;
    if (l1 > bv) { bv = l1; bt = 1; }   // strict > : first max (matches jnp.argmax)
    if (l2 > bv) { bv = l2; bt = 2; }
    if (l3 > bv) { bv = l3; bt = 3; }
    float4 g;
    g.x = (bt == 0) ? 1.f : 0.f;
    g.y = (bt == 1) ? 1.f : 0.f;
    g.z = (bt == 2) ? 1.f : 0.f;
    g.w = (bt == 3) ? 1.f : 0.f;
    *(float4*)(gate_out + (size_t)n * NTILES) = g;
    int slot = atomicAdd(&counts[bt], 1);
    perm[bt * N + slot] = n;
  }
}

// ---------- phase 1.5: weight f32 -> f16 conversion into ws ---------------
// WupH: same row-major [DFF][C]. WdnH compacted: [t][256][1024] where
// WdnH[t][r][k] = Wdn[(t*256+r)*DFF + t*1024 + k] (the only blocks routing touches).

__global__ void convert_weights_kernel(const float* __restrict__ Wup,
                                       const float* __restrict__ Wdn,
                                       _Float16* __restrict__ WupH,
                                       _Float16* __restrict__ WdnH) {
  int idx = blockIdx.x * blockDim.x + threadIdx.x;   // one float4 per thread
  int total4 = (WUP_ELEMS + WDN_ELEMS) / 4;
  if (idx >= total4) return;
  float4 f;
  _Float16* dst;
  if (idx < WUP_ELEMS / 4) {
    f = *(const float4*)(Wup + (size_t)idx * 4);
    dst = WupH + (size_t)idx * 4;
  } else {
    size_t j = ((size_t)idx - WUP_ELEMS / 4) * 4;    // element idx into WdnH
    int t = (int)(j >> 18);                          // / (256*1024)
    int rem = (int)(j & ((256 * 1024) - 1));
    int r = rem >> 10, k = rem & 1023;
    f = *(const float4*)(Wdn + (size_t)(t * 256 + r) * DFF + t * C_DIM + k);
    dst = WdnH + j;
  }
  h4 hv;
  hv[0] = (_Float16)f.x; hv[1] = (_Float16)f.y;
  hv[2] = (_Float16)f.z; hv[3] = (_Float16)f.w;
  *(h4*)dst = hv;
}

// ---------- phase 2: routed fused FFN with WMMA ----------------------------
// Block = 256 threads (8 waves). Each block: one (tile t, 64-row group).
// F16W: B-operands from pre-converted f16 weights in ws.

template <bool F16W>
__global__ void __launch_bounds__(256)
ffn_kernel(const float* __restrict__ x,
           const float* __restrict__ Wup,
           const _Float16* __restrict__ WupH,
           const float* __restrict__ bup,
           const float* __restrict__ Wdn,
           const _Float16* __restrict__ WdnH,
           const float* __restrict__ bdn,
           float* __restrict__ out,
           const int* __restrict__ counts,
           const int* __restrict__ perm, int N) {
  extern __shared__ char smem[];
  _Float16* xlds   = (_Float16*)smem;
  _Float16* hlds   = (_Float16*)(smem + HLDS_OFF);
  float*    olds   = (float*)(smem + OLDS_OFF);       // aliases dead xlds
  int*      rowidx = (int*)(smem + RIDX_OFF);

  int t  = blockIdx.x & 3;
  int mb = blockIdx.x >> 2;
  int cnt = counts[t];
  if (mb * BM >= cnt) return;   // uniform early exit (fixed worst-case grid)

  int tid = threadIdx.x;
  int wave = tid >> 5, lane = tid & 31;
  int n16 = lane & 15, half = lane >> 4;

  if (tid < BM) {
    int slot = mb * BM + tid;
    rowidx[tid] = (slot < cnt) ? perm[t * N + slot] : -1;
  }
  __syncthreads();

  // ---- gather BM x-rows into LDS as f16 (8 rows per wave) ----
#pragma unroll 1
  for (int rr = 0; rr < BM / 8; rr++) {
    int m = wave * (BM / 8) + rr;
    int r = rowidx[m];
    const float4* src = (const float4*)(x + (size_t)(r < 0 ? 0 : r) * C_DIM);
#pragma unroll 1
    for (int j = lane; j < C_DIM / 4; j += 32) {
      float4 f;
      if (r >= 0) f = src[j];
      else        f.x = f.y = f.z = f.w = 0.f;
      h4 hv;
      hv[0] = (_Float16)f.x; hv[1] = (_Float16)f.y;
      hv[2] = (_Float16)f.z; hv[3] = (_Float16)f.w;
      *(h4*)(xlds + (size_t)m * XSTR + j * 4) = hv;
    }
  }
  __syncthreads();

  const _Float16* ax0 = xlds + (size_t)(0 * 16 + n16) * XSTR;
  const _Float16* ax1 = xlds + (size_t)(1 * 16 + n16) * XSTR;
  const _Float16* ax2 = xlds + (size_t)(2 * 16 + n16) * XSTR;
  const _Float16* ax3 = xlds + (size_t)(3 * 16 + n16) * XSTR;

  // ---- stage B: H = relu(X @ Wup_t^T + bup_t) ----
#pragma unroll 1
  for (int ci = 0; ci < 8; ci++) {
    int chunk = wave + 8 * ci;          // 0..63
    int col = chunk * 16 + n16;         // 0..1023 within tile
    float bb = bup[t * C_DIM + col];
    v8f c0, c1, c2, c3;
#pragma unroll
    for (int i = 0; i < 8; i++) { c0[i] = bb; c1[i] = bb; c2[i] = bb; c3[i] = bb; }
    const float*    wrow  = Wup  + (size_t)(t * C_DIM + col) * C_DIM;
    const _Float16* wrowh = WupH + (size_t)(t * C_DIM + col) * C_DIM;
#pragma unroll 2
    for (int kb = 0; kb < C_DIM; kb += 32) {
      v16h b = F16W ? load_b_frag_f16(wrowh + kb + half * 16)
                    : load_b_frag_f32(wrow  + kb + half * 16);
      v16h f0 = load_a_frag(ax0, kb, half);
      v16h f1 = load_a_frag(ax1, kb, half);
      v16h f2 = load_a_frag(ax2, kb, half);
      v16h f3 = load_a_frag(ax3, kb, half);
      c0 = WMMA_F16(f0, b, c0);
      c1 = WMMA_F16(f1, b, c1);
      c2 = WMMA_F16(f2, b, c2);
      c3 = WMMA_F16(f3, b, c3);
    }
#pragma unroll
    for (int i = 0; i < 8; i++) {
      int rowi = i + 8 * half;
      float v0 = c0[i] > 0.f ? c0[i] : 0.f;
      float v1 = c1[i] > 0.f ? c1[i] : 0.f;
      float v2 = c2[i] > 0.f ? c2[i] : 0.f;
      float v3 = c3[i] > 0.f ? c3[i] : 0.f;
      hlds[(size_t)(rowi +  0) * XSTR + col] = (_Float16)v0;
      hlds[(size_t)(rowi + 16) * XSTR + col] = (_Float16)v1;
      hlds[(size_t)(rowi + 32) * XSTR + col] = (_Float16)v2;
      hlds[(size_t)(rowi + 48) * XSTR + col] = (_Float16)v3;
    }
  }
  __syncthreads();

  const _Float16* ah0 = hlds + (size_t)(0 * 16 + n16) * XSTR;
  const _Float16* ah1 = hlds + (size_t)(1 * 16 + n16) * XSTR;
  const _Float16* ah2 = hlds + (size_t)(2 * 16 + n16) * XSTR;
  const _Float16* ah3 = hlds + (size_t)(3 * 16 + n16) * XSTR;

  // ---- stage C: OUT_t = H @ Wdn_t^T + bdn_t (olds aliases dead xlds) ----
#pragma unroll 1
  for (int ci = 0; ci < 2; ci++) {
    int chunk = wave + 8 * ci;          // 0..15
    int ccol = chunk * 16 + n16;        // 0..255 within 256-wide out tile
    float bb = bdn[t * 256 + ccol];
    v8f c0, c1, c2, c3;
#pragma unroll
    for (int i = 0; i < 8; i++) { c0[i] = bb; c1[i] = bb; c2[i] = bb; c3[i] = bb; }
    const float*    wrow  = Wdn  + (size_t)(t * 256 + ccol) * DFF + t * C_DIM;
    const _Float16* wrowh = WdnH + (size_t)(t * 256 + ccol) * C_DIM;   // compacted
#pragma unroll 2
    for (int kb = 0; kb < C_DIM; kb += 32) {
      v16h b = F16W ? load_b_frag_f16(wrowh + kb + half * 16)
                    : load_b_frag_f32(wrow  + kb + half * 16);
      v16h f0 = load_a_frag(ah0, kb, half);
      v16h f1 = load_a_frag(ah1, kb, half);
      v16h f2 = load_a_frag(ah2, kb, half);
      v16h f3 = load_a_frag(ah3, kb, half);
      c0 = WMMA_F16(f0, b, c0);
      c1 = WMMA_F16(f1, b, c1);
      c2 = WMMA_F16(f2, b, c2);
      c3 = WMMA_F16(f3, b, c3);
    }
#pragma unroll
    for (int i = 0; i < 8; i++) {
      int rowi = i + 8 * half;
      olds[(rowi +  0) * 256 + ccol] = c0[i];
      olds[(rowi + 16) * 256 + ccol] = c1[i];
      olds[(rowi + 32) * 256 + ccol] = c2[i];
      olds[(rowi + 48) * 256 + ccol] = c3[i];
    }
  }
  __syncthreads();

  // ---- write full output rows: computed 256-tile + explicit zeros elsewhere ----
#pragma unroll 1
  for (int p = tid; p < BM * 256; p += 256) {
    int m = p >> 8;
    int r = rowidx[m];
    if (r < 0) continue;
    int col = (p & 255) * 4;
    int rel = col - t * 256;
    float4 v;
    if (rel >= 0 && rel < 256) v = *(float4*)(olds + m * 256 + rel);
    else { v.x = v.y = v.z = v.w = 0.f; }
    *(float4*)(out + (size_t)r * C_DIM + col) = v;
  }
}

// ---------- launch ---------------------------------------------------------

extern "C" void kernel_launch(void* const* d_in, const int* in_sizes, int n_in,
                              void* d_out, int out_size, void* d_ws, size_t ws_size,
                              hipStream_t stream) {
  const float* x   = (const float*)d_in[0];
  const float* Wg  = (const float*)d_in[1];
  const float* bg  = (const float*)d_in[2];
  const float* Wup = (const float*)d_in[3];
  const float* bup = (const float*)d_in[4];
  const float* Wdn = (const float*)d_in[5];
  const float* bdn = (const float*)d_in[6];
  float* out = (float*)d_out;

  int N = in_sizes[0] / C_DIM;              // B*T rows

  char* ws = (char*)d_ws;
  int* counts = (int*)ws;
  int* perm   = (int*)(ws + WS_PERM_OFF);
  size_t perm_bytes = (size_t)NTILES * N * sizeof(int);
  size_t wuph_off = WS_PERM_OFF + ((perm_bytes + 255) & ~(size_t)255);
  size_t wdnh_off = wuph_off + (size_t)WUP_ELEMS * 2;
  size_t ws_need  = wdnh_off + (size_t)WDN_ELEMS * 2;
  bool f16w = (ws_size >= ws_need);
  _Float16* WupH = (_Float16*)(ws + wuph_off);
  _Float16* WdnH = (_Float16*)(ws + wdnh_off);

  float* gate_out = out + (size_t)N * C_DIM;

  hipLaunchKernelGGL(zero_counts_kernel, dim3(1), dim3(32), 0, stream, counts);
  hipLaunchKernelGGL(gate_kernel, dim3(N / 8), dim3(256), 0, stream,
                     x, Wg, bg, gate_out, counts, perm, N);

  dim3 grid(NTILES * (N / BM));
  if (f16w) {
    int total4 = (WUP_ELEMS + WDN_ELEMS) / 4;
    hipLaunchKernelGGL(convert_weights_kernel, dim3((total4 + 255) / 256), dim3(256),
                       0, stream, Wup, Wdn, WupH, WdnH);
    hipLaunchKernelGGL((ffn_kernel<true>), grid, dim3(256), LDS_TOTAL, stream,
                       x, Wup, WupH, bup, Wdn, WdnH, bdn, out, counts, perm, N);
  } else {
    hipLaunchKernelGGL((ffn_kernel<false>), grid, dim3(256), LDS_TOTAL, stream,
                       x, Wup, WupH, bup, Wdn, WdnH, bdn, out, counts, perm, N);
  }
}
